// VectorQuantizer_76871324664189
// MI455X (gfx1250) — matrix operand order repaired and checked by reading, not compiled, not run on hardware
//
#include <hip/hip_runtime.h>
#include <hip/hip_bf16.h>
#include <math.h>

typedef __attribute__((ext_vector_type(16))) _Float16 v16h;
typedef __attribute__((ext_vector_type(8)))  float    v8f;

#define NUM_CODES   1024
#define DIM         128
#define N_TOTAL     73728          // 2 * 16 * 48 * 48
#define SPATIAL     36864          // 16 * 48 * 48 (per-batch spatial)
#define M_PER_WG    128
#define NUM_WG      (N_TOTAL / M_PER_WG)   // 576

// d_out layout (reference tuple, flattened):
#define OUT_Q     0
#define OUT_LOSS  9437184
#define OUT_PERP  9437185
#define OUT_IDX   9437186

// d_ws layout:
#define WS_CBH    0          // _Float16[1024*128]  (256 KB)
#define WS_ENORM  262144     // float[1024]
#define WS_COUNTS 266240     // float[1024]
#define WS_LOSS   270336     // float[1]

__global__ void vq_zero(float* counts, float* lossSum) {
    int i = blockIdx.x * blockDim.x + threadIdx.x;
    if (i < NUM_CODES) counts[i] = 0.0f;
    if (i == 0) *lossSum = 0.0f;
}

// One block per code: f32 codebook row -> f16 copy + ||e||^2
__global__ __launch_bounds__(128) void vq_prep(const float* __restrict__ cb,
                                               _Float16* __restrict__ cbh,
                                               float* __restrict__ eNorm) {
    int code = blockIdx.x;
    int t = threadIdx.x;                 // 0..127
    float v = cb[code * DIM + t];
    cbh[code * DIM + t] = (_Float16)v;
    float s = v * v;
    #pragma unroll
    for (int d = 16; d >= 1; d >>= 1) s += __shfl_xor(s, d, 32);
    __shared__ float r[4];
    if ((t & 31) == 0) r[t >> 5] = s;
    __syncthreads();
    if (t == 0) eNorm[code] = r[0] + r[1] + r[2] + r[3];
}

__global__ __launch_bounds__(256) void vq_main(const float* __restrict__ x,
                                               const float* __restrict__ cbf,
                                               const _Float16* __restrict__ cbh,
                                               const float* __restrict__ eNorm,
                                               float* __restrict__ counts,
                                               float* __restrict__ lossSum,
                                               float* __restrict__ out) {
    // A-matrix fragments (16x32 f16, k = 32*kchunk + 16*(lane/16) + half): [wave][kchunk][lane][16 halves]
    __shared__ alignas(32) _Float16 Afrag[8][4][32][16];   // 32 KB
    // Codebook block: 128 codes x 128 k, row-major f16 == B-fragment layout
    __shared__ alignas(32) _Float16 Bblk[128 * 128];       // 32 KB
    __shared__ int   idx_sh[M_PER_WG];
    __shared__ float red[8];

    const int t    = threadIdx.x;
    const int wave = t >> 5;
    const int lane = t & 31;
    const int ln   = lane & 15;
    const int hi   = lane >> 4;

    const int n0 = blockIdx.x * M_PER_WG;
    const int b  = n0 / SPATIAL;
    const int s0 = n0 % SPATIAL;
    const float* xbase = x + (size_t)b * (DIM * (size_t)SPATIAL) + s0;

    // ---- Load X tile (coalesced) -> f16, swizzled into A-fragment layout ----
    {
        const int j  = t & 127;      // position within WG tile
        const int ch = t >> 7;       // 0/1
        const int wv = j >> 4;
        const int m  = j & 15;
        #pragma unroll 4
        for (int it = 0; it < 64; ++it) {
            int c = it * 2 + ch;                       // channel == K index
            float v = xbase[(size_t)c * SPATIAL + j];
            int kc = c >> 5;
            int kl = c & 31;
            int fl = m + 16 * (kl >> 4);               // fragment lane
            int e  = kl & 15;                          // element in v16h
            Afrag[wv][kc][fl][e] = (_Float16)v;
        }
    }

    // Prefetch codebook block 0 into registers (overlaps with LDS swizzle/barrier)
    const uint4* gsrc = (const uint4*)cbh;             // 2048 uint4 per 128-code block
    uint4 pre[8];
    #pragma unroll
    for (int i = 0; i < 8; ++i) pre[i] = gsrc[t + 256 * i];

    __syncthreads();

    // Per-wave A fragments in registers
    v16h a0 = *(const v16h*)&Afrag[wave][0][lane][0];
    v16h a1 = *(const v16h*)&Afrag[wave][1][lane][0];
    v16h a2 = *(const v16h*)&Afrag[wave][2][lane][0];
    v16h a3 = *(const v16h*)&Afrag[wave][3][lane][0];

    float best[8];
    int   bidx[8];
    #pragma unroll
    for (int v = 0; v < 8; ++v) { best[v] = 3.4e38f; bidx[v] = 0; }

    // ---- Sweep codebook in 8 blocks of 128 codes (software pipelined) ----
    for (int cblk = 0; cblk < 8; ++cblk) {
        __syncthreads();
        {   // commit prefetched block to LDS
            uint4* dst = (uint4*)Bblk;
            #pragma unroll
            for (int i = 0; i < 8; ++i) dst[t + 256 * i] = pre[i];
        }
        __syncthreads();

        // issue next block's global loads now; they complete under the WMMA work
        if (cblk < 7) {
            const uint4* nsrc = gsrc + (size_t)(cblk + 1) * 2048;
            #pragma unroll
            for (int i = 0; i < 8; ++i) pre[i] = nsrc[t + 256 * i];
        }

        // preload this block's ||e||^2 values (one per tile for this lane)
        float en8[8];
        #pragma unroll
        for (int tile = 0; tile < 8; ++tile)
            en8[tile] = eNorm[cblk * 128 + tile * 16 + ln];

        #pragma unroll 4
        for (int tile = 0; tile < 8; ++tile) {
            const v16h* bp = (const v16h*)&Bblk[(tile * 16 + ln) * DIM + hi * 16];
            v8f acc = {};
            acc = __builtin_amdgcn_wmma_f32_16x16x32_f16(false, a0, false, bp[0], (short)0, acc, false, false);
            acc = __builtin_amdgcn_wmma_f32_16x16x32_f16(false, a1, false, bp[2], (short)0, acc, false, false);
            acc = __builtin_amdgcn_wmma_f32_16x16x32_f16(false, a2, false, bp[4], (short)0, acc, false, false);
            acc = __builtin_amdgcn_wmma_f32_16x16x32_f16(false, a3, false, bp[6], (short)0, acc, false, false);

            int code = cblk * 128 + tile * 16 + ln;    // this lane's codebook column
            float en = en8[tile];
            #pragma unroll
            for (int v = 0; v < 8; ++v) {
                float d = en - 2.0f * acc[v];          // ||e||^2 - 2 x.e
                if (d < best[v]) { best[v] = d; bidx[v] = code; }
            }
        }
    }

    // ---- Cross-lane argmin within each 16-lane half (C/D layout: n = lane&15) ----
    #pragma unroll
    for (int d = 1; d < 16; d <<= 1) {
        #pragma unroll
        for (int v = 0; v < 8; ++v) {
            float ob = __shfl_xor(best[v], d, 32);
            int   oi = __shfl_xor(bidx[v], d, 32);
            if (ob < best[v] || (ob == best[v] && oi < bidx[v])) {
                best[v] = ob; bidx[v] = oi;
            }
        }
    }
    if (ln == 0) {   // lanes 0 (m=0..7) and 16 (m=8..15) of each wave
        #pragma unroll
        for (int v = 0; v < 8; ++v)
            idx_sh[wave * 16 + hi * 8 + v] = bidx[v];
    }
    __syncthreads();

    // ---- Indices out + histogram ----
    if (t < M_PER_WG) {
        int idx = idx_sh[t];
        out[OUT_IDX + n0 + t] = (float)idx;
        atomicAdd(&counts[idx], 1.0f);
    }

    // ---- Quantized output (coalesced, transposed) + loss partial ----
    float lsum = 0.0f;
    {
        const int j  = t & 127;
        const int ch = t >> 7;
        const int code = idx_sh[j];
        float* obase = out + OUT_Q + (size_t)b * (DIM * (size_t)SPATIAL) + s0;
        #pragma unroll 4
        for (int it = 0; it < 64; ++it) {
            int c = it * 2 + ch;
            float q  = cbf[code * DIM + c];            // L2-hot gather (512 KB)
            float xv = xbase[(size_t)c * SPATIAL + j];
            float dd = q - xv;
            lsum += dd * dd;
            obase[(size_t)c * SPATIAL + j] = q;        // straight-through == q
        }
    }
    #pragma unroll
    for (int d = 16; d >= 1; d >>= 1) lsum += __shfl_xor(lsum, d, 32);
    if (lane == 0) red[wave] = lsum;
    __syncthreads();
    if (t == 0) {
        float tot = 0.0f;
        #pragma unroll
        for (int i = 0; i < 8; ++i) tot += red[i];
        atomicAdd(lossSum, tot);
    }
}

__global__ __launch_bounds__(256) void vq_finalize(const float* __restrict__ counts,
                                                   const float* __restrict__ lossSum,
                                                   float* __restrict__ out) {
    int t = threadIdx.x;
    float s = 0.0f;
    for (int i = t; i < NUM_CODES; i += 256) {
        float p = counts[i] * (1.0f / (float)N_TOTAL);
        s += p * logf(p + 1e-10f);
    }
    #pragma unroll
    for (int d = 16; d >= 1; d >>= 1) s += __shfl_xor(s, d, 32);
    __shared__ float r[8];
    if ((t & 31) == 0) r[t >> 5] = s;
    __syncthreads();
    if (t == 0) {
        float tot = 0.0f;
        #pragma unroll
        for (int i = 0; i < 8; ++i) tot += r[i];
        out[OUT_LOSS] = 1.25f * (*lossSum) / (float)((size_t)N_TOTAL * DIM);
        out[OUT_PERP] = expf(-tot);
    }
}

extern "C" void kernel_launch(void* const* d_in, const int* in_sizes, int n_in,
                              void* d_out, int out_size, void* d_ws, size_t ws_size,
                              hipStream_t stream) {
    const float* x  = (const float*)d_in[0];   // (2,128,16,48,48) fp32
    const float* cb = (const float*)d_in[1];   // (1024,128) fp32
    float* out = (float*)d_out;
    char*  ws  = (char*)d_ws;

    _Float16* cbh    = (_Float16*)(ws + WS_CBH);
    float*    eNorm  = (float*)(ws + WS_ENORM);
    float*    counts = (float*)(ws + WS_COUNTS);
    float*    lossS  = (float*)(ws + WS_LOSS);

    vq_zero<<<(NUM_CODES + 255) / 256, 256, 0, stream>>>(counts, lossS);
    vq_prep<<<NUM_CODES, 128, 0, stream>>>(cb, cbh, eNorm);
    vq_main<<<NUM_WG, 256, 0, stream>>>(x, cb, cbh, eNorm, counts, lossS, out);
    vq_finalize<<<1, 256, 0, stream>>>(counts, lossS, out);
}
